// MLPPredictor_88201448391221
// MI455X (gfx1250) — matrix operand order repaired and checked by reading, not compiled, not run on hardware
//
#include <hip/hip_runtime.h>
#include <hip/hip_bf16.h>

typedef __attribute__((ext_vector_type(16))) _Float16 v16h;
typedef __attribute__((ext_vector_type(8)))  _Float16 v8h;
typedef __attribute__((ext_vector_type(4)))  _Float16 v4h;
typedef __attribute__((ext_vector_type(8)))  float    v8f;
typedef __attribute__((ext_vector_type(4)))  float    v4f;

#define GNN_N_NODES 100000
#define GNN_N_EDGES 1000000
#define GNN_D 128
#define GNN_C 86

static constexpr int TILES = GNN_N_EDGES / 16;      // 62500 tiles of 16 edges (even)
static constexpr int WPB   = 8;                     // waves per block (256 thr)
static constexpr int TPW   = 8;                     // tiles per wave (processed in pairs)
static constexpr int GRID  = (TILES + WPB * TPW - 1) / (WPB * TPW); // 977

// ---- workspace layout (bytes) ----
static constexpr size_t H16_HALVES = (size_t)GNN_N_NODES * GNN_D;   // 12.8M
static constexpr size_t W1T_HALVES = 64 * 512;                      // 8 kk * 8 nt blocks
static constexpr size_t W2T_HALVES = 24 * 512;                      // 4 kk * 6 nt blocks
static constexpr size_t H16_OFF = 0;
static constexpr size_t W1T_OFF = H16_OFF + H16_HALVES * sizeof(_Float16);
static constexpr size_t W2T_OFF = W1T_OFF + W1T_HALVES * sizeof(_Float16);

// ---- LDS layout (halves) ----
static constexpr int SW1_HALVES = (int)W1T_HALVES;          // 64 KB
static constexpr int SW2_HALVES = (int)W2T_HALVES;          // 24 KB
static constexpr int HID_STRIDE = 136;                      // 128 + 8 pad (16B-aligned rows, de-banked)
static constexpr int SHID_PER_TILE = 16 * HID_STRIDE;       // hid scratch per 16-edge tile
static constexpr size_t SMEM_BYTES =
    (size_t)(SW1_HALVES + SW2_HALVES + WPB * 2 * SHID_PER_TILE) * sizeof(_Float16); // ~156 KB

// ------------------------------------------------------------------
// Kernel 0: convert node features f32 -> f16 (vector of 4 per thread)
// ------------------------------------------------------------------
__global__ void cvt_h_kernel(const float* __restrict__ h, _Float16* __restrict__ h16) {
    size_t i = ((size_t)blockIdx.x * blockDim.x + threadIdx.x) * 4;
    v4f f = __builtin_nontemporal_load((const v4f*)(h + i));
    v4h o;
    o[0] = (_Float16)f[0]; o[1] = (_Float16)f[1];
    o[2] = (_Float16)f[2]; o[3] = (_Float16)f[3];
    *(v4h*)(h16 + i) = o;   // regular store: h16 should stay L2-resident
}

// ------------------------------------------------------------------
// Kernel 1: re-layout W1/W2 into WMMA B-fragment order, f32 -> f16.
// Fragment block b holds (kk, nt); within a block: lane l (0..31) owns
// 16 contiguous halves:  K = kk*32 + (l<16?0:16) + j,  N = nt*16 + (l&15).
// ------------------------------------------------------------------
__global__ void cvt_w_kernel(const float* __restrict__ W1, const float* __restrict__ W2,
                             _Float16* __restrict__ w1t, _Float16* __restrict__ w2t) {
    int idx = blockIdx.x * blockDim.x + threadIdx.x;   // 0 .. 45055
    if (idx < (int)W1T_HALVES) {
        int b  = idx >> 9;              // 0..63
        int l  = (idx >> 4) & 31;
        int j  = idx & 15;
        int kk = b >> 3;                // 0..7
        int nt = b & 7;                 // 0..7
        int K  = kk * 32 + ((l < 16) ? 0 : 16) + j;   // 0..255
        int N  = nt * 16 + (l & 15);                  // 0..127
        w1t[idx] = (_Float16)W1[K * GNN_D + N];
    } else {
        int i2 = idx - (int)W1T_HALVES;  // 0 .. 12287
        int b  = i2 >> 9;                // 0..23
        int l  = (i2 >> 4) & 31;
        int j  = i2 & 15;
        int kk = b / 6;                  // 0..3
        int nt = b % 6;                  // 0..5
        int K  = kk * 32 + ((l < 16) ? 0 : 16) + j;   // 0..127
        int N  = nt * 16 + (l & 15);                  // 0..95 (pad >=86 with 0)
        w2t[i2] = (N < GNN_C) ? (_Float16)W2[K * GNN_C + N] : (_Float16)0.0f;
    }
}

// ------------------------------------------------------------------
// Main fused kernel, 2 edge-tiles per wave so each LDS B-fragment
// feeds two WMMAs (LDS bandwidth is the binding resource).
// ------------------------------------------------------------------
__global__ void __launch_bounds__(256)
mlp_edge_kernel(const _Float16* __restrict__ h16,
                const _Float16* __restrict__ w1t,
                const _Float16* __restrict__ w2t,
                const float* __restrict__ b1,
                const float* __restrict__ b2,
                const long long* __restrict__ src,
                const long long* __restrict__ dst,
                float* __restrict__ out) {
    extern __shared__ _Float16 smem[];
    _Float16* sW1 = smem;
    _Float16* sW2 = smem + SW1_HALVES;

    const int tid = threadIdx.x;

    // ---- stage fragment-layout weights into LDS (flat 16B-chunk copy) ----
    {
        const int nChunks  = (SW1_HALVES + SW2_HALVES) / 8;   // 5632
        const int w1Chunks = SW1_HALVES / 8;
        for (int c = tid; c < nChunks; c += blockDim.x) {
            v8h ch;
            if (c < w1Chunks) ch = *(const v8h*)(w1t + (size_t)c * 8);
            else              ch = *(const v8h*)(w2t + (size_t)(c - w1Chunks) * 8);
            *(v8h*)(smem + (size_t)c * 8) = ch;
        }
    }
    __syncthreads();

    const int lane    = tid & 31;
    const int wave    = tid >> 5;
    const int m       = lane & 15;        // row within 16-edge tile
    const int halfSel = lane >> 4;        // 0 | 1
    const int kbase   = halfSel * 16;     // A-fragment K sub-offset
    const int mbase   = halfSel * 8;      // C-layout M offset

    _Float16* sHid0 = smem + SW1_HALVES + SW2_HALVES + wave * 2 * SHID_PER_TILE;
    _Float16* sHid1 = sHid0 + SHID_PER_TILE;

    // per-lane bias splats (C-layout: lane owns column n for all 8 rows)
    float bias1[8];
#pragma unroll
    for (int nt = 0; nt < 8; ++nt) bias1[nt] = b1[nt * 16 + m];
    float bias2[6];
#pragma unroll
    for (int nt = 0; nt < 6; ++nt) {
        int n = nt * 16 + m;
        bias2[nt] = (n < GNN_C) ? b2[n] : 0.0f;
    }

    const long long tile0 = ((long long)blockIdx.x * WPB + wave) * TPW;

    for (int p = 0; p < TPW / 2; ++p) {
        const long long tileA = tile0 + 2 * p;
        if (tileA >= TILES) break;          // TILES is even -> tileA+1 also valid

        const long long eA = tileA * 16 + m;
        const long long eB = eA + 16;
        const _Float16* upA = h16 + src[eA] * GNN_D;
        const _Float16* vpA = h16 + dst[eA] * GNN_D;
        const _Float16* upB = h16 + src[eB] * GNN_D;
        const _Float16* vpB = h16 + dst[eB] * GNN_D;

        // ---------------- layer 1: [2x 16x256] @ [256x128] ----------------
        v8f acc[2][8];
#pragma unroll
        for (int nt = 0; nt < 8; ++nt) {
#pragma unroll
            for (int r = 0; r < 8; ++r) { acc[0][nt][r] = bias1[nt]; acc[1][nt][r] = bias1[nt]; }
        }

#pragma unroll
        for (int kk = 0; kk < 8; ++kk) {
            const _Float16* rpA = (kk < 4) ? (upA + kk * 32) : (vpA + (kk - 4) * 32);
            const _Float16* rpB = (kk < 4) ? (upB + kk * 32) : (vpB + (kk - 4) * 32);
            union { v16h v; v8h h[2]; } A0, A1;
            A0.h[0] = *(const v8h*)(rpA + kbase);
            A0.h[1] = *(const v8h*)(rpA + kbase + 8);
            A1.h[0] = *(const v8h*)(rpB + kbase);
            A1.h[1] = *(const v8h*)(rpB + kbase + 8);
            const _Float16* wb = sW1 + (size_t)(kk * 8) * 512 + lane * 16;
#pragma unroll
            for (int nt = 0; nt < 8; ++nt) {
                v16h B = *(const v16h*)(wb + nt * 512);   // loaded once, used twice
                acc[0][nt] = __builtin_amdgcn_wmma_f32_16x16x32_f16(
                    false, A0.v, false, B, (short)0, acc[0][nt], false, false);
                acc[1][nt] = __builtin_amdgcn_wmma_f32_16x16x32_f16(
                    false, A1.v, false, B, (short)0, acc[1][nt], false, false);
            }
        }

        // ---- ReLU + f16, transpose through per-wave LDS into A-layout ----
#pragma unroll
        for (int nt = 0; nt < 8; ++nt) {
            int n = nt * 16 + m;
#pragma unroll
            for (int r = 0; r < 8; ++r) {
                float x0 = acc[0][nt][r]; x0 = (x0 > 0.0f) ? x0 : 0.0f;
                float x1 = acc[1][nt][r]; x1 = (x1 > 0.0f) ? x1 : 0.0f;
                sHid0[(mbase + r) * HID_STRIDE + n] = (_Float16)x0;
                sHid1[(mbase + r) * HID_STRIDE + n] = (_Float16)x1;
            }
        }
        asm volatile("" ::: "memory");   // keep DS stores before DS loads (HW in-order per wave)

        // ---------------- layer 2: [2x 16x128] @ [128x96] ----------------
        v8f acc2[2][6];
#pragma unroll
        for (int nt = 0; nt < 6; ++nt) {
#pragma unroll
            for (int r = 0; r < 8; ++r) { acc2[0][nt][r] = bias2[nt]; acc2[1][nt][r] = bias2[nt]; }
        }

#pragma unroll
        for (int kk = 0; kk < 4; ++kk) {
            const _Float16* hp0 = sHid0 + m * HID_STRIDE + kk * 32;
            const _Float16* hp1 = sHid1 + m * HID_STRIDE + kk * 32;
            union { v16h v; v8h h[2]; } A0, A1;
            A0.h[0] = *(const v8h*)(hp0 + kbase);
            A0.h[1] = *(const v8h*)(hp0 + kbase + 8);
            A1.h[0] = *(const v8h*)(hp1 + kbase);
            A1.h[1] = *(const v8h*)(hp1 + kbase + 8);
            const _Float16* wb = sW2 + (size_t)(kk * 6) * 512 + lane * 16;
#pragma unroll
            for (int nt = 0; nt < 6; ++nt) {
                v16h B = *(const v16h*)(wb + nt * 512);   // loaded once, used twice
                acc2[0][nt] = __builtin_amdgcn_wmma_f32_16x16x32_f16(
                    false, A0.v, false, B, (short)0, acc2[0][nt], false, false);
                acc2[1][nt] = __builtin_amdgcn_wmma_f32_16x16x32_f16(
                    false, A1.v, false, B, (short)0, acc2[1][nt], false, false);
            }
        }
        asm volatile("" ::: "memory");   // next iteration rewrites sHid

        // ---------------- store scores (nontemporal: pure 344MB stream) ----
        const long long rowA = tileA * 16;
        const long long rowB = rowA + 16;
#pragma unroll
        for (int nt = 0; nt < 6; ++nt) {
            int n = nt * 16 + m;
            if (n < GNN_C) {
#pragma unroll
                for (int r = 0; r < 8; ++r) {
                    __builtin_nontemporal_store(acc2[0][nt][r],
                        out + (rowA + mbase + r) * (long long)GNN_C + n);
                    __builtin_nontemporal_store(acc2[1][nt][r],
                        out + (rowB + mbase + r) * (long long)GNN_C + n);
                }
            }
        }
    }
}

extern "C" void kernel_launch(void* const* d_in, const int* in_sizes, int n_in,
                              void* d_out, int out_size, void* d_ws, size_t ws_size,
                              hipStream_t stream) {
    const float*     h   = (const float*)d_in[0];
    const float*     W1  = (const float*)d_in[1];
    const float*     b1  = (const float*)d_in[2];
    const float*     W2  = (const float*)d_in[3];
    const float*     b2  = (const float*)d_in[4];
    const long long* src = (const long long*)d_in[5];
    const long long* dst = (const long long*)d_in[6];
    float* out = (float*)d_out;

    _Float16* h16 = (_Float16*)((char*)d_ws + H16_OFF);
    _Float16* w1t = (_Float16*)((char*)d_ws + W1T_OFF);
    _Float16* w2t = (_Float16*)((char*)d_ws + W2T_OFF);

    // 12.8M elems / 4 per thread / 256 per block = 12500 blocks (exact)
    cvt_h_kernel<<<12500, 256, 0, stream>>>(h, h16);
    // 45056 halves total = 176 * 256 (exact)
    cvt_w_kernel<<<176, 256, 0, stream>>>(W1, W2, w1t, w2t);

    mlp_edge_kernel<<<GRID, 256, SMEM_BYTES, stream>>>(h16, w1t, w2t, b1, b2, src, dst, out);
}
